// HybridHydroTransformer_58858231824484
// MI455X (gfx1250) — compile-verified, matrix operand order checked by reading, not run on hardware
//
#include <hip/hip_runtime.h>
#include <math.h>

typedef __bf16 bf16;
typedef __attribute__((ext_vector_type(16))) __bf16 v16bf;
typedef __attribute__((ext_vector_type(8)))  __bf16 v8bf;
typedef __attribute__((ext_vector_type(8)))  float  v8f;

namespace {

constexpr int NPIX   = 256 * 256;     // H*W
constexpr int TSTEPS = 365;
constexpr int WARM   = 100;
constexpr int SOUT   = TSTEPS - WARM; // 265
constexpr int CIN    = 8;
constexpr int HID    = 256;
constexpr int NF     = 7;
constexpr int KP1    = 32;            // layer-1 K padded 8 -> 32
constexpr int NP3    = 16;            // layer-3 N padded 7 -> 16

constexpr float SM0v = 50.0f, GW0v = 100.0f, MELTv = 3.0f;

constexpr int WAVES = 8;
constexpr int BLOCK = WAVES * 32;     // 256 threads
constexpr int PIXPB = WAVES * 16;     // 128 pixels per block
constexpr int NBLK  = NPIX / PIXPB;   // 512 blocks

// ---- LDS layout (bytes); all offsets 16B aligned ----
constexpr int OFF_W1 = 0;                          // 256x32 bf16  (W1^T, K padded)
constexpr int OFF_W2 = OFF_W1 + HID * KP1 * 2;     // 256x256 bf16 (W2^T)
constexpr int OFF_W3 = OFF_W2 + HID * HID * 2;     // 16x256 bf16  (W3^T, N padded)
constexpr int OFF_B1 = OFF_W3 + NP3 * HID * 2;     // 256 f32
constexpr int OFF_B2 = OFF_B1 + HID * 4;           // 256 f32
constexpr int OFF_B3 = OFF_B2 + HID * 4;           // 16  f32
constexpr int OFF_X  = OFF_B3 + NP3 * 4;           // per-wave 16x32 bf16 input stage
constexpr int OFF_H  = OFF_X + WAVES * 16 * KP1 * 2; // per-wave 16x256 bf16 hidden stage
constexpr int OFF_F  = OFF_H + WAVES * 16 * HID * 2; // per-wave 16x16 f32 flux stage
constexpr int LDSB   = OFF_F + WAVES * 16 * NP3 * 4; // ~234 KB (WGP has 320 KB)

__device__ __forceinline__ v8f wmma_bf16(v16bf a, v16bf b, v8f c) {
  return __builtin_amdgcn_wmma_f32_16x16x32_bf16(false, a, false, b,
                                                 (short)0, c, false, false);
}

// Load one 16x32 (rows x K) bf16 operand tile from row-major LDS storage into
// the WMMA A/B per-lane layout: lanes 0-15 hold K {0..7,16..23} of row=lane,
// lanes 16-31 hold K {8..15,24..31} of row=lane-16. Two ds_load_b128 per lane.
__device__ __forceinline__ v16bf load_tile(const bf16* base, int stride,
                                           int kbase, int lane) {
  const int row = lane & 15;
  const int hi  = (lane >> 4) & 1;
  const bf16* p = base + row * stride + kbase + hi * 8;
  union { v16bf v; v8bf h[2]; } u;
  u.h[0] = *(const v8bf*)(p);       // K: kbase + (hi?8:0) .. +7
  u.h[1] = *(const v8bf*)(p + 16);  // K: kbase + 16 + (hi?8:0) .. +7
  return u.v;
}

__device__ __forceinline__ float softplusf(float x) {
  return fmaxf(x, 0.0f) + log1pf(expf(-fabsf(x)));
}

__global__ void init_ploss_kernel(float* p) {
  int i = threadIdx.x + blockIdx.x * blockDim.x;
  if (i < SOUT) p[i] = 0.0f;
}

__global__ __launch_bounds__(BLOCK)
void hydro_scan_kernel(const float* __restrict__ pr, const float* __restrict__ tm,
                       const float* __restrict__ pe, const float* __restrict__ la,
                       const float* __restrict__ W1g, const float* __restrict__ b1g,
                       const float* __restrict__ W2g, const float* __restrict__ b2g,
                       const float* __restrict__ W3g, const float* __restrict__ b3g,
                       float* __restrict__ out) {
  extern __shared__ char smem[];
  bf16*  lW1 = (bf16*)(smem + OFF_W1);
  bf16*  lW2 = (bf16*)(smem + OFF_W2);
  bf16*  lW3 = (bf16*)(smem + OFF_W3);
  float* lb1 = (float*)(smem + OFF_B1);
  float* lb2 = (float*)(smem + OFF_B2);
  float* lb3 = (float*)(smem + OFF_B3);
  bf16*  lX  = (bf16*)(smem + OFF_X);
  bf16*  lH  = (bf16*)(smem + OFF_H);
  float* lF  = (float*)(smem + OFF_F);

  const int tid = threadIdx.x;

  // ---- one-time: transpose weights into LDS as bf16 (resident for 365 steps)
  for (int idx = tid; idx < HID * KP1; idx += BLOCK) {   // W1t[n][k]
    int n = idx / KP1, k = idx % KP1;
    lW1[idx] = (bf16)((k < CIN) ? W1g[k * HID + n] : 0.0f);
  }
  for (int idx = tid; idx < HID * HID; idx += BLOCK) {   // W2t[n][k] = W2[k][n]
    int k = idx / HID, n = idx % HID;                    // coalesced global read
    lW2[n * HID + k] = (bf16)W2g[idx];
  }
  for (int idx = tid; idx < NP3 * HID; idx += BLOCK) {   // W3t[n][k]
    int n = idx / HID, k = idx % HID;
    lW3[idx] = (bf16)((n < NF) ? W3g[k * NF + n] : 0.0f);
  }
  for (int i = tid; i < HID; i += BLOCK) { lb1[i] = b1g[i]; lb2[i] = b2g[i]; }
  if (tid < NP3) lb3[tid] = (tid < NF) ? b3g[tid] : 0.0f;
  for (int i = tid; i < WAVES * 16 * KP1; i += BLOCK) lX[i] = (bf16)0.0f; // K-pad zeros persist
  __syncthreads();

  const int wave  = tid >> 5;
  const int lane  = tid & 31;
  const int nlane = lane & 15;
  const int hi8   = (lane >> 4) << 3;            // 0 or 8 (C-layout M offset)
  const int pixbase = blockIdx.x * PIXPB + wave * 16;
  const int pix     = pixbase + nlane;           // this lane's pixel (lanes 0-15)

  bf16*  xst = lX + wave * 16 * KP1;
  bf16*  hst = lH + wave * 16 * HID;
  float* fst = lF + wave * 16 * NP3;

  // per-pixel recurrent state in registers (lanes 0-15)
  float snow = 0.0f, interc = 0.0f, soil = SM0v, gw = GW0v;

  const size_t SG = (size_t)SOUT * NPIX;
  float* o_run   = out;
  float* o_ploss = out + SG;
  float* o_snow  = o_ploss + SOUT;
  float* o_soil  = o_snow + SG;
  float* o_gw    = o_soil + SG;
  float* o_int   = o_gw + SG;
  float* o_aet   = o_int + SG;
  float* o_q1    = o_aet + SG;
  float* o_q2    = o_q1 + SG;
  float* o_q3    = o_q2 + SG;

  for (int t = 0; t < TSTEPS; ++t) {
    float wos = 0.0f, petv = 0.0f;
    const size_t gidx = (size_t)t * NPIX + pix;

    if (lane < 16) {
      float precip = pr[gidx], temp = tm[gidx], lai = la[gidx];
      petv = pe[gidx];
      if (t + 1 < TSTEPS) {  // global_prefetch_b8 for next step's fields
        __builtin_prefetch(pr + gidx + NPIX, 0, 1);
        __builtin_prefetch(tm + gidx + NPIX, 0, 1);
        __builtin_prefetch(pe + gidx + NPIX, 0, 1);
        __builtin_prefetch(la + gidx + NPIX, 0, 1);
      }
      float rain = (temp > 0.0f) ? precip : 0.0f;
      snow += precip - rain;
      float melt = fminf(snow, fmaxf(temp, 0.0f) * MELTv);
      snow -= melt;
      wos = rain + melt;
      bf16* xr = xst + nlane * KP1;              // row-major 16x32, cols 8..31 zero
      xr[0] = (bf16)wos;    xr[1] = (bf16)petv; xr[2] = (bf16)temp; xr[3] = (bf16)snow;
      xr[4] = (bf16)interc; xr[5] = (bf16)soil; xr[6] = (bf16)gw;   xr[7] = (bf16)lai;
    }
    // same-wave DS ops are in-order: stores above are visible to loads below.

    v8f acc[16];

    // ---- layer 1: x(16x8 pad32) @ W1 -> 16x256, relu, stage as bf16 ----
    {
      v16bf a1 = load_tile(xst, KP1, 0, lane);
      #pragma unroll
      for (int n = 0; n < 16; ++n) {
        float bv = lb1[n * 16 + nlane];          // N depends only on lane in C-layout
        v8f c;
        #pragma unroll
        for (int r = 0; r < 8; ++r) c[r] = bv;
        v16bf b = load_tile(lW1 + (n * 16) * KP1, KP1, 0, lane);
        acc[n] = wmma_bf16(a1, b, c);
      }
      #pragma unroll
      for (int n = 0; n < 16; ++n)
        #pragma unroll
        for (int r = 0; r < 8; ++r)
          hst[(r + hi8) * HID + n * 16 + nlane] = (bf16)fmaxf(acc[n][r], 0.0f);
    }

    // ---- layer 2: h1(16x256) @ W2 -> 16x256 (94% of FLOPs) ----
    #pragma unroll
    for (int n = 0; n < 16; ++n) {
      float bv = lb2[n * 16 + nlane];
      #pragma unroll
      for (int r = 0; r < 8; ++r) acc[n][r] = bv;
    }
    for (int k = 0; k < HID / 32; ++k) {
      v16bf a = load_tile(hst, HID, k * 32, lane);   // A reused across 16 N-tiles
      #pragma unroll
      for (int n = 0; n < 16; ++n) {
        v16bf b = load_tile(lW2 + (n * 16) * HID, HID, k * 32, lane);
        acc[n] = wmma_bf16(a, b, acc[n]);
      }
    }
    #pragma unroll
    for (int n = 0; n < 16; ++n)
      #pragma unroll
      for (int r = 0; r < 8; ++r)
        hst[(r + hi8) * HID + n * 16 + nlane] = (bf16)fmaxf(acc[n][r], 0.0f);

    // ---- layer 3: h2(16x256) @ W3 -> 16x16 (N padded), softplus ----
    {
      v8f acc3;
      float bv = lb3[nlane];
      #pragma unroll
      for (int r = 0; r < 8; ++r) acc3[r] = bv;
      for (int k = 0; k < HID / 32; ++k) {
        v16bf a = load_tile(hst, HID, k * 32, lane);
        v16bf b = load_tile(lW3, HID, k * 32, lane);
        acc3 = wmma_bf16(a, b, acc3);
      }
      #pragma unroll
      for (int r = 0; r < 8; ++r)
        fst[(r + hi8) * NP3 + nlane] = softplusf(acc3[r]);
    }

    // ---- elementwise water balance + outputs (lanes 0-15, one pixel each) ----
    float pl = 0.0f;
    if (lane < 16) {
      const float* fr = fst + nlane * NP3;
      float q1d = fr[0], q2d = fr[1], q3d = fr[2], infd = fr[3];
      float perd = fr[4], aetd = fr[5], capd = fr[6];

      float aetc = fminf(aetd, petv);
      float raet = fminf(aetc, interc + soil);
      float sf   = fminf(1.0f, wos / (q1d + infd + 1e-9f));
      float q1   = q1d * sf;
      float inf  = infd * sf;
      float ti = interc + wos - inf - q1 - aetc;
      float ts = soil + inf - perd - q2d + capd;
      float tg = gw + perd - q3d - capd;
      float na = fmaxf(-ti, 0.0f), nb = fmaxf(-ts, 0.0f), nc = fmaxf(-tg, 0.0f);
      pl = na * na + nb * nb + nc * nc;
      interc = fmaxf(ti, 0.0f);
      soil   = fmaxf(ts, 0.0f);
      gw     = fmaxf(tg, 0.0f);

      if (t >= WARM) {
        size_t o = (size_t)(t - WARM) * NPIX + pix;
        o_run[o]  = q1d + q2d + q3d;
        o_snow[o] = snow;
        o_soil[o] = soil;
        o_gw[o]   = gw;
        o_int[o]  = interc;
        o_aet[o]  = raet;
        o_q1[o]   = q1;
        o_q2[o]   = q2d;
        o_q3[o]   = q3d;
      }
    }
    if (t >= WARM) {
      #pragma unroll
      for (int s = 16; s > 0; s >>= 1) pl += __shfl_xor(pl, s, 32);
      if (lane == 0) atomicAdd(o_ploss + (t - WARM), pl * (1.0f / NPIX));
    }
  }
}

}  // namespace

extern "C" void kernel_launch(void* const* d_in, const int* in_sizes, int n_in,
                              void* d_out, int out_size, void* d_ws, size_t ws_size,
                              hipStream_t stream) {
  (void)in_sizes; (void)n_in; (void)out_size; (void)d_ws; (void)ws_size;
  const float* pr = (const float*)d_in[0];
  const float* tm = (const float*)d_in[1];
  const float* pe = (const float*)d_in[2];
  const float* la = (const float*)d_in[3];
  const float* W1 = (const float*)d_in[4];
  const float* b1 = (const float*)d_in[5];
  const float* W2 = (const float*)d_in[6];
  const float* b2 = (const float*)d_in[7];
  const float* W3 = (const float*)d_in[8];
  const float* b3 = (const float*)d_in[9];
  float* out = (float*)d_out;

  // Allow ~234 KB dynamic LDS (WGP has 320 KB on CDNA5). Idempotent; errors ignored.
  (void)hipFuncSetAttribute((const void*)hydro_scan_kernel,
                            hipFuncAttributeMaxDynamicSharedMemorySize, LDSB);

  // ploss is accumulated via atomics -> zero its 265 slots first (same stream).
  init_ploss_kernel<<<1, 512, 0, stream>>>(out + (size_t)SOUT * NPIX);

  hydro_scan_kernel<<<NBLK, BLOCK, LDSB, stream>>>(pr, tm, pe, la,
                                                   W1, b1, W2, b2, W3, b3, out);
}
